// MultiHeadAttention_32779190403390
// MI455X (gfx1250) — compile-verified, matrix operand order checked by reading, not run on hardware
//
#include <hip/hip_runtime.h>

// ---------------------------------------------------------------------------
// Types
// ---------------------------------------------------------------------------
typedef __bf16 bf16_t;
typedef __attribute__((ext_vector_type(16))) __bf16 v16bf;
typedef __attribute__((ext_vector_type(8)))  __bf16 v8bf;
typedef __attribute__((ext_vector_type(4)))  __bf16 v4bf;
typedef __attribute__((ext_vector_type(8)))  float  v8f;

// Problem constants (from reference): B=4, T=2048, C=1024, H=16, D=64
#define BATCH 4
#define SEQ   2048
#define CDIM  1024
#define NHEAD 16
#define HDIM  64
#define ROWS  (BATCH * SEQ)           // 8192
#define QKVC  (3 * CDIM)              // 3072

// ---------------------------------------------------------------------------
// f32 -> bf16 (round to nearest even)
// ---------------------------------------------------------------------------
__device__ __forceinline__ unsigned short f2u16(float f) {
  union { float f; unsigned u; } v; v.f = f;
  return (unsigned short)((v.u + 0x7FFFu + ((v.u >> 16) & 1u)) >> 16);
}
__device__ __forceinline__ bf16_t f2bf(float f) {
  union { unsigned short u; bf16_t b; } c; c.u = f2u16(f); return c.b;
}

// Build a 16-wide bf16 fragment from two aligned 16-byte chunks.
// Matches CDNA5 16-bit A/B fragment layout: comps 0..7 and 8..15 are the two
// contiguous 8-element K-groups selected per lane-half.
__device__ __forceinline__ v16bf frag16(const bf16_t* p0, const bf16_t* p1) {
  v8bf lo = *(const v8bf*)p0;
  v8bf hi = *(const v8bf*)p1;
  v16bf r;
#pragma unroll
  for (int i = 0; i < 8; ++i) { r[i] = lo[i]; r[i + 8] = hi[i]; }
  return r;
}

__device__ __forceinline__ void lds_fence() {
  asm volatile("s_wait_dscnt 0" ::: "memory");
  __builtin_amdgcn_wave_barrier();
}

// ---------------------------------------------------------------------------
// Kernel 1: elementwise f32 -> bf16 (4 elems/thread)
// ---------------------------------------------------------------------------
__global__ __launch_bounds__(256) void k_cvt(const float* __restrict__ s,
                                             bf16_t* __restrict__ d, int n4) {
  int i = blockIdx.x * blockDim.x + threadIdx.x;
  if (i >= n4) return;
  float4 v = ((const float4*)s)[i];
  ushort4 o;
  o.x = f2u16(v.x); o.y = f2u16(v.y); o.z = f2u16(v.z); o.w = f2u16(v.w);
  *((ushort4*)(d + 4 * (size_t)i)) = o;
}

// ---------------------------------------------------------------------------
// Kernel 2: bf16 GEMM  C[M,N] = A[M,K] * B[K,N], f32 accumulate via WMMA.
// Block: 256 threads = 8 waves; block tile 128x64; wave tile 32x32 (2x2 WMMA).
// LDS double-buffered: tile k+1 is fetched into registers while tile k is
// consumed, hiding global latency behind WMMA. Pitch 40 (80B) => conflict-free
// 16B-aligned ds_load_b128 fragment reads.
// ---------------------------------------------------------------------------
#define LDK 40

template <int OUTF32>
__global__ __launch_bounds__(256)
void k_gemm(const bf16_t* __restrict__ A, const bf16_t* __restrict__ Bm,
            bf16_t* __restrict__ obf, float* __restrict__ of32,
            int M, int N, int K) {
  __shared__ bf16_t lA[2][128 * LDK];   // [m][k], padded
  __shared__ bf16_t lB[2][64 * LDK];    // [n][k] (B transposed), padded

  const int tid  = threadIdx.x;
  const int lane = tid & 31;
  const int w    = tid >> 5;
  const int lm   = lane & 15;
  const int lh   = lane >> 4;
  const int wr   = w >> 1;           // 0..3
  const int wc   = w & 1;            // 0..1
  const int mBase = blockIdx.y * 128;
  const int nBase = blockIdx.x * 64;

  // Global->LDS mappings
  const int aRow = tid >> 2;          // 0..63
  const int aCol = (tid & 3) * 8;     // 0,8,16,24
  const int bN   = (tid & 15) * 4;    // 0..60
  const int bK   = tid >> 4;          // 0..15

  v8f acc[2][2] = {};
  v8bf aReg[2];
  v4bf bReg[2];

  auto loadTiles = [&](int k0) {
#pragma unroll
    for (int p = 0; p < 2; ++p)
      aReg[p] =
          *(const v8bf*)(A + (size_t)(mBase + p * 64 + aRow) * K + k0 + aCol);
#pragma unroll
    for (int p = 0; p < 2; ++p)
      bReg[p] =
          *(const v4bf*)(Bm + (size_t)(k0 + p * 16 + bK) * N + nBase + bN);
  };
  auto storeTiles = [&](int buf) {
#pragma unroll
    for (int p = 0; p < 2; ++p)
      *(v8bf*)(lA[buf] + (p * 64 + aRow) * LDK + aCol) = aReg[p];
#pragma unroll
    for (int p = 0; p < 2; ++p) {
      int kk = p * 16 + bK;
#pragma unroll
      for (int e = 0; e < 4; ++e)
        lB[buf][(bN + e) * LDK + kk] = bReg[p][e];
    }
  };

  loadTiles(0);
  storeTiles(0);
  __syncthreads();

  for (int k0 = 0; k0 < K; k0 += 32) {
    const int buf = (k0 >> 5) & 1;
    const bool more = (k0 + 32) < K;
    if (more) loadTiles(k0 + 32);   // global fetch overlaps WMMAs below

    v16bf af[2], bfr[2];
#pragma unroll
    for (int i = 0; i < 2; ++i) {
      int row = wr * 32 + i * 16 + lm;
      af[i]  = frag16(lA[buf] + row * LDK + lh * 8,
                      lA[buf] + row * LDK + 16 + lh * 8);
      int col = wc * 32 + i * 16 + lm;
      bfr[i] = frag16(lB[buf] + col * LDK + lh * 8,
                      lB[buf] + col * LDK + 16 + lh * 8);
    }
#pragma unroll
    for (int i = 0; i < 2; ++i)
#pragma unroll
      for (int j = 0; j < 2; ++j)
        acc[i][j] = __builtin_amdgcn_wmma_f32_16x16x32_bf16(
            false, af[i], false, bfr[j], (short)0, acc[i][j], false, false);

    if (more) storeTiles(buf ^ 1);
    __syncthreads();
  }

  // C/D layout: lane(0..15)=N col, VGPR r => M row r (+8 for upper half lanes)
#pragma unroll
  for (int i = 0; i < 2; ++i)
#pragma unroll
    for (int j = 0; j < 2; ++j) {
      int col = nBase + wc * 32 + j * 16 + lm;
#pragma unroll
      for (int r = 0; r < 8; ++r) {
        int row = mBase + wr * 32 + i * 16 + lh * 8 + r;
        float v = acc[i][j][r];
        if (OUTF32) of32[(size_t)row * N + col] = v;
        else        obf[(size_t)row * N + col] = f2bf(v);
      }
    }
}

// ---------------------------------------------------------------------------
// Kernel 3: causal flash attention. One wave per 16-query tile; 4 waves/block
// with private LDS slices (no inter-wave sync -> divergent causal trip counts
// are safe). Per 32-key block: all K/V global loads are issued up front so
// they clause and overlap the S WMMAs; V's LDS transpose retires under the
// softmax expf chain. S = Q*K^T (4 WMMAs) then O += P*V (4 WMMAs).
// qkv: [B*T, 3C] bf16 (q | k | v per row). y: [B*T, C] bf16, head-merged.
// ---------------------------------------------------------------------------
__global__ __launch_bounds__(128)
void k_attn(const bf16_t* __restrict__ qkv, bf16_t* __restrict__ y) {
  __shared__ bf16_t lV[4][HDIM * LDK];  // [d][k] transposed V block
  __shared__ float  lS[4][16 * 36];     // S scores, padded pitch 36
  __shared__ bf16_t lP[4][16 * LDK];    // P in A-fragment staging layout
  __shared__ float  lStat[4][48];       // [0..15]=m  [16..31]=l  [32..47]=alpha

  const int tid  = threadIdx.x;
  const int wv   = tid >> 5;
  const int lane = tid & 31;
  const int lm   = lane & 15;
  const int lh   = lane >> 4;

  bf16_t* Vs = lV[wv];
  float*  Ss = lS[wv];
  bf16_t* Ps = lP[wv];
  float*  St = lStat[wv];

  const int tile = blockIdx.x * 4 + wv;            // 0..8191
  const int bh   = tile >> 7;                      // / (T/16)
  const int qt   = tile & 127;
  const int b    = bh >> 4;
  const int h    = bh & 15;
  const int qb   = qt * 16;
  const float scale = 0.125f;                      // 1/sqrt(64)

  // Q fragments (A-matrix 16x32, two k-steps covering D=64), read once.
  v16bf qf[2];
  {
    const bf16_t* qp = qkv + ((size_t)(b * SEQ) + qb + lm) * QKVC + h * HDIM;
    qf[0] = frag16(qp + lh * 8,      qp + 16 + lh * 8);
    qf[1] = frag16(qp + 32 + lh * 8, qp + 48 + lh * 8);
  }

  if (lane < 16) { St[lm] = -1e30f; St[16 + lm] = 0.0f; }
  lds_fence();

  v8f o[4] = {};

  const int kEnd = qb + 16;                        // max key (exclusive)
  for (int kb = 0; kb < kEnd; kb += 32) {
    // ---- issue V global loads early (independent of S/softmax) -----------
    v8bf vreg[8];
    {
      const bf16_t* vp =
          qkv + ((size_t)(b * SEQ) + kb + lane) * QKVC + 2 * CDIM + h * HDIM;
#pragma unroll
      for (int d8 = 0; d8 < 8; ++d8) vreg[d8] = *(const v8bf*)(vp + d8 * 8);
    }

    // ---- load ALL K fragments, then S = Q*K^T (loads clause together) ----
    v16bf kf[2][2];
#pragma unroll
    for (int j = 0; j < 2; ++j) {
      int key = kb + j * 16 + lm;
      const bf16_t* kp =
          qkv + ((size_t)(b * SEQ) + key) * QKVC + CDIM + h * HDIM;
      kf[j][0] = frag16(kp + lh * 8,      kp + 16 + lh * 8);
      kf[j][1] = frag16(kp + 32 + lh * 8, kp + 48 + lh * 8);
    }
    v8f s[2] = {};
#pragma unroll
    for (int j = 0; j < 2; ++j) {
      s[j] = __builtin_amdgcn_wmma_f32_16x16x32_bf16(
          false, qf[0], false, kf[j][0], (short)0, s[j], false, false);
      s[j] = __builtin_amdgcn_wmma_f32_16x16x32_bf16(
          false, qf[1], false, kf[j][1], (short)0, s[j], false, false);
    }

    // spill S to LDS in [row][key] order
#pragma unroll
    for (int j = 0; j < 2; ++j)
#pragma unroll
      for (int r = 0; r < 8; ++r)
        Ss[(lh * 8 + r) * 36 + j * 16 + lm] = s[j][r];
    lds_fence();

    // ---- V transpose store to LDS; retires under softmax VALU below ------
#pragma unroll
    for (int d8 = 0; d8 < 8; ++d8)
#pragma unroll
      for (int e = 0; e < 8; ++e)
        Vs[(d8 * 8 + e) * LDK + lane] = vreg[d8][e];

    // ---- streaming softmax: lanes 0..15 each own one query row -----------
    if (lane < 16) {
      const int t  = qb + lm;
      float mold = St[lm];
      float rmax = mold;
      for (int kk = 0; kk < 32; ++kk) {
        int key = kb + kk;
        float v = (key <= t) ? Ss[lm * 36 + kk] * scale : -1e30f;
        rmax = v > rmax ? v : rmax;
      }
      float alpha = __expf(mold - rmax);
      float sum = 0.0f;
      for (int kk = 0; kk < 32; ++kk) {
        int key = kb + kk;
        float v = (key <= t) ? Ss[lm * 36 + kk] * scale : -1e30f;
        float p = __expf(v - rmax);
        sum += p;
        Ps[lm * LDK + kk] = f2bf(p);
      }
      St[16 + lm] = St[16 + lm] * alpha + sum;
      St[lm]      = rmax;
      St[32 + lm] = alpha;
    }
    lds_fence();

    // ---- rescale running O by alpha(row) ----------------------------------
#pragma unroll
    for (int r = 0; r < 8; ++r) {
      float a = St[32 + lh * 8 + r];
#pragma unroll
      for (int n = 0; n < 4; ++n) o[n][r] *= a;
    }

    // ---- O += P * V (4 N-tiles of 16 dims) --------------------------------
    v16bf pf = frag16(Ps + lm * LDK + lh * 8, Ps + lm * LDK + 16 + lh * 8);
#pragma unroll
    for (int n = 0; n < 4; ++n) {
      int dcol = n * 16 + lm;
      v16bf vf = frag16(Vs + dcol * LDK + lh * 8,
                        Vs + dcol * LDK + 16 + lh * 8);
      o[n] = __builtin_amdgcn_wmma_f32_16x16x32_bf16(
          false, pf, false, vf, (short)0, o[n], false, false);
    }
    lds_fence();
  }

  // ---- normalize and store y[b, t, h*64 + d] ------------------------------
#pragma unroll
  for (int r = 0; r < 8; ++r) {
    int m = lh * 8 + r;
    float inv = 1.0f / St[16 + m];
    int t = qb + m;
    bf16_t* yp = y + ((size_t)(b * SEQ) + t) * CDIM + h * HDIM;
#pragma unroll
    for (int n = 0; n < 4; ++n)
      yp[n * 16 + lm] = f2bf(o[n][r] * inv);
  }
}

// ---------------------------------------------------------------------------
// Launch
// ---------------------------------------------------------------------------
extern "C" void kernel_launch(void* const* d_in, const int* in_sizes, int n_in,
                              void* d_out, int out_size, void* d_ws,
                              size_t ws_size, hipStream_t stream) {
  const float* x      = (const float*)d_in[0];
  const float* w_attn = (const float*)d_in[1];
  const float* w_proj = (const float*)d_in[2];
  float* out = (float*)d_out;

  // Workspace layout (bf16 buffers), ~92 MB total
  char* w = (char*)d_ws;
  bf16_t* xb   = (bf16_t*)w; w += (size_t)ROWS * CDIM * 2;
  bf16_t* wab  = (bf16_t*)w; w += (size_t)CDIM * QKVC * 2;
  bf16_t* wpb  = (bf16_t*)w; w += (size_t)CDIM * CDIM * 2;
  bf16_t* qkvb = (bf16_t*)w; w += (size_t)ROWS * QKVC * 2;
  bf16_t* yb   = (bf16_t*)w; w += (size_t)ROWS * CDIM * 2;

  // 1) convert inputs to bf16
  k_cvt<<<(ROWS * CDIM / 4) / 256, 256, 0, stream>>>(x, xb, ROWS * CDIM / 4);
  k_cvt<<<(CDIM * QKVC / 4) / 256, 256, 0, stream>>>(w_attn, wab,
                                                     CDIM * QKVC / 4);
  k_cvt<<<(CDIM * CDIM / 4) / 256, 256, 0, stream>>>(w_proj, wpb,
                                                     CDIM * CDIM / 4);

  // 2) qkv = x @ w_attn   [8192,1024]x[1024,3072] -> bf16
  {
    dim3 g(QKVC / 64, ROWS / 128);
    k_gemm<0><<<g, 256, 0, stream>>>(xb, wab, qkvb, nullptr, ROWS, QKVC, CDIM);
  }

  // 3) causal flash attention -> yb [8192,1024] bf16
  {
    int tiles = BATCH * NHEAD * (SEQ / 16);   // 8192
    k_attn<<<tiles / 4, 128, 0, stream>>>(qkvb, yb);
  }

  // 4) out = y @ w_proj   [8192,1024]x[1024,1024] -> f32 d_out
  {
    dim3 g(CDIM / 64, ROWS / 128);
    k_gemm<1><<<g, 256, 0, stream>>>(yb, wpb, nullptr, out, ROWS, CDIM, CDIM);
  }
}
